// GroupedQueryAttention_11484742549591
// MI455X (gfx1250) — matrix operand order repaired and checked by reading, not compiled, not run on hardware
//
#include <hip/hip_runtime.h>

// ---------------------------------------------------------------------------
// GQA attention block for gfx1250 (MI455X), bf16 WMMA with fp32 accumulate.
//   B=2, S=2048, HIDDEN=2048, H=32, Hkv=8, D=64, GROUPS=4, ROPE base 5e5
// ---------------------------------------------------------------------------

typedef __attribute__((ext_vector_type(16))) __bf16 v16bf;
typedef __attribute__((ext_vector_type(8)))  __bf16 v8bf;
typedef __attribute__((ext_vector_type(8)))  float  v8f;

#define SEQ   2048
#define HID   2048
#define NH    32
#define NKV   8
#define HD    64
#define NEGI  (-1e9f)

__device__ __forceinline__ __bf16 f2bf(float f) {
    unsigned u = __builtin_bit_cast(unsigned, f);
    unsigned r = u + 0x7FFFu + ((u >> 16) & 1u);   // round-to-nearest-even
    unsigned short h = (unsigned short)(r >> 16);
    return __builtin_bit_cast(__bf16, h);
}

__device__ __forceinline__ v8f vzero8() {
    v8f z;
#pragma unroll
    for (int i = 0; i < 8; ++i) z[i] = 0.0f;
    return z;
}

__device__ __forceinline__ v8f wmma_bf16(v16bf a, v16bf b, v8f c) {
    // D = A(16x32 bf16) * B(32x16 bf16) + C(16x16 f32)
    return __builtin_amdgcn_wmma_f32_16x16x32_bf16(
        false, a, false, b, (short)0, c, false, false);
}

// ---------------------------------------------------------------------------
// Kernel 0: weight transpose + f32->bf16 convert.  in[K,N] f32 -> out[N,K] bf16
// ---------------------------------------------------------------------------
__global__ __launch_bounds__(256) void wtc_kernel(const float* __restrict__ in,
                                                  __bf16* __restrict__ outp,
                                                  int K, int N) {
    int idx = blockIdx.x * 256 + threadIdx.x;
    int total = K * N;
    if (idx >= total) return;
    int n = idx / K;
    int k = idx - n * K;
    outp[idx] = f2bf(in[(size_t)k * N + n]);
}

// ---------------------------------------------------------------------------
// Kernel 1: projection GEMM (+ optional RoPE, optional transposed store).
//   X  [4096, 2048] f32   (rows = b*S+s tokens)
//   Wt [nheads*64, 2048] bf16  (transposed weight)
//   out bf16:
//     do_transpose==0 : [b, h, s, 64]
//     do_transpose==1 : [b, h, 64, s]
// One wave computes a 16-row x 64-col (one head) tile.
// ---------------------------------------------------------------------------
__global__ __launch_bounds__(32) void proj_kernel(const float*  __restrict__ X,
                                                  const __bf16* __restrict__ Wt,
                                                  __bf16* __restrict__ outp,
                                                  int do_rope, int do_transpose) {
    const int lane   = threadIdx.x & 31;
    const int laneLo = lane & 15;
    const int laneHi = lane >> 4;
    const int rtile  = blockIdx.x;            // 256 tiles of 16 rows
    const int h      = blockIdx.y;            // head
    const int nheads = gridDim.y;
    const int row    = rtile * 16 + laneLo;   // A-matrix row for this lane

    v8f acc[4];
#pragma unroll
    for (int t = 0; t < 4; ++t) acc[t] = vzero8();

    const float* xr = X + (size_t)row * HID;
    const int kaOff = laneHi ? 16 : 0;        // A-layout half-lane K offset
    const int kbOff = laneHi ? 16 : 0;        // B-layout half-lane K offset

    for (int kb = 0; kb < HID; kb += 32) {
        // ---- A tile (16x32 bf16) from f32 activations ----
        v16bf a;
        const float4* x4 = reinterpret_cast<const float4*>(xr + kb + kaOff);
        float4 c0 = x4[0], c1 = x4[1];        // K = k0 .. k0+7
        float4 c2 = x4[2], c3 = x4[3];        // K = k0+8 .. k0+15
        a[0]=f2bf(c0.x); a[1]=f2bf(c0.y); a[2]=f2bf(c0.z); a[3]=f2bf(c0.w);
        a[4]=f2bf(c1.x); a[5]=f2bf(c1.y); a[6]=f2bf(c1.z); a[7]=f2bf(c1.w);
        a[8]=f2bf(c2.x); a[9]=f2bf(c2.y); a[10]=f2bf(c2.z); a[11]=f2bf(c2.w);
        a[12]=f2bf(c3.x); a[13]=f2bf(c3.y); a[14]=f2bf(c3.z); a[15]=f2bf(c3.w);

        // ---- 4 B tiles (32x16 bf16), contiguous per lane from Wt ----
#pragma unroll
        for (int t = 0; t < 4; ++t) {
            const __bf16* wp = Wt + (size_t)(h * 64 + t * 16 + laneLo) * HID
                                  + kb + kbOff;
            v16bf bm = *reinterpret_cast<const v16bf*>(wp);
            acc[t] = wmma_bf16(a, bm, acc[t]);
        }
    }

    // ---- epilogue: RoPE (per-lane, since a head's d and d+-32 share a lane)
#pragma unroll
    for (int v = 0; v < 8; ++v) {
        int gr = rtile * 16 + v + 8 * laneHi;     // global token row
        int b  = gr >> 11;
        int s  = gr & (SEQ - 1);
        float res[4];
        if (do_rope) {
#pragma unroll
            for (int t2 = 0; t2 < 2; ++t2) {
                int dmod = t2 * 16 + laneLo;      // frequency index (d % 32)
                float inv = __powf(500000.0f, -(float)dmod * (1.0f / 32.0f));
                float th  = (float)s * inv;
                float cs  = __cosf(th);
                float sn  = __sinf(th);
                res[t2]     = acc[t2][v] * cs - acc[t2 + 2][v] * sn;
                res[t2 + 2] = acc[t2 + 2][v] * cs + acc[t2][v] * sn;
            }
        } else {
#pragma unroll
            for (int t = 0; t < 4; ++t) res[t] = acc[t][v];
        }
#pragma unroll
        for (int t = 0; t < 4; ++t) {
            int d = t * 16 + laneLo;
            size_t off;
            if (do_transpose)
                off = ((size_t)(b * nheads + h) * HD + d) * SEQ + s;   // [b,h,d,s]
            else
                off = ((size_t)(b * nheads + h) * SEQ + s) * HD + d;   // [b,h,s,d]
            outp[off] = f2bf(res[t]);
        }
    }
}

// ---------------------------------------------------------------------------
// Kernel 2: flash attention, one wave per (b, h, 16-query tile).
//   Qr [b,32,s,64] bf16 (rope'd), Kr [b,8,s,64] bf16 (rope'd), Vt [b,8,64,s]
//   AO [b, s, 32*64] bf16
// ---------------------------------------------------------------------------
__global__ __launch_bounds__(32) void attn_kernel(const __bf16* __restrict__ Qr,
                                                  const __bf16* __restrict__ Kr,
                                                  const __bf16* __restrict__ Vt,
                                                  __bf16* __restrict__ AO) {
    __shared__ __bf16 pl[16][32];            // P transpose bounce (1 KB)

    const int lane   = threadIdx.x & 31;
    const int laneLo = lane & 15;
    const int laneHi = lane >> 4;
    const int qt  = blockIdx.x;              // 0..127 query tiles
    const int h   = blockIdx.y;              // 0..31
    const int b   = blockIdx.z;              // 0..1
    const int hkv = h >> 2;                  // GQA: 4 query heads per kv head
    const int ka  = laneHi ? 16 : 0;

    // ---- Q as two A-matrices (d 0..31 and d 32..63) ----
    const __bf16* qp = Qr + ((size_t)(b * NH + h) * SEQ + qt * 16 + laneLo) * HD;
    v16bf qa0, qa1;
    {
        v8bf lo = *reinterpret_cast<const v8bf*>(qp + ka);
        v8bf hi = *reinterpret_cast<const v8bf*>(qp + ka + 8);
#pragma unroll
        for (int i = 0; i < 8; ++i) { qa0[i] = lo[i]; qa0[8 + i] = hi[i]; }
        lo = *reinterpret_cast<const v8bf*>(qp + 32 + ka);
        hi = *reinterpret_cast<const v8bf*>(qp + 32 + ka + 8);
#pragma unroll
        for (int i = 0; i < 8; ++i) { qa1[i] = lo[i]; qa1[8 + i] = hi[i]; }
    }

    v8f o[4];
#pragma unroll
    for (int t = 0; t < 4; ++t) o[t] = vzero8();
    float m[8], l[8];
#pragma unroll
    for (int v = 0; v < 8; ++v) { m[v] = -3.0e38f; l[v] = 0.0f; }

    const __bf16* kbaseP = Kr + (size_t)(b * NKV + hkv) * SEQ * HD;
    const __bf16* vbaseP = Vt + (size_t)(b * NKV + hkv) * HD * SEQ;
    const int nkb = (qt * 16 + 16 + 31) >> 5;     // causal: key blocks needed

    for (int kblk = 0; kblk < nkb; ++kblk) {
        const int k0 = kblk * 32;

        // ---- scores: S[16x32] = Q[16x64] @ K^T, 4 WMMAs ----
        const __bf16* kp0 = kbaseP + (size_t)(k0 + laneLo) * HD + (laneHi ? 16 : 0);
        const __bf16* kp1 = kp0 + (size_t)16 * HD;
        v16bf kb00 = *reinterpret_cast<const v16bf*>(kp0);        // keys 0-15, d 0-31
        v16bf kb01 = *reinterpret_cast<const v16bf*>(kp0 + 32);   // keys 0-15, d 32-63
        v16bf kb10 = *reinterpret_cast<const v16bf*>(kp1);        // keys 16-31
        v16bf kb11 = *reinterpret_cast<const v16bf*>(kp1 + 32);
        v8f s0 = wmma_bf16(qa0, kb00, vzero8());
        s0 = wmma_bf16(qa1, kb01, s0);
        v8f s1 = wmma_bf16(qa0, kb10, vzero8());
        s1 = wmma_bf16(qa1, kb11, s1);

        // ---- online softmax, per C-layout row slot ----
#pragma unroll
        for (int v = 0; v < 8; ++v) {
            int row = qt * 16 + v + 8 * laneHi;
            int c0  = k0 + laneLo;
            int c1  = c0 + 16;
            float x0 = s0[v] * 0.125f + (c0 > row ? NEGI : 0.0f);
            float x1 = s1[v] * 0.125f + (c1 > row ? NEGI : 0.0f);
            float rmax = fmaxf(x0, x1);
            rmax = fmaxf(rmax, __shfl_xor(rmax, 1, 32));
            rmax = fmaxf(rmax, __shfl_xor(rmax, 2, 32));
            rmax = fmaxf(rmax, __shfl_xor(rmax, 4, 32));
            rmax = fmaxf(rmax, __shfl_xor(rmax, 8, 32));
            float mnew = fmaxf(m[v], rmax);
            float corr = __expf(m[v] - mnew);
            float p0 = __expf(x0 - mnew);
            float p1 = __expf(x1 - mnew);
            float rsum = p0 + p1;
            rsum += __shfl_xor(rsum, 1, 32);
            rsum += __shfl_xor(rsum, 2, 32);
            rsum += __shfl_xor(rsum, 4, 32);
            rsum += __shfl_xor(rsum, 8, 32);
            l[v] = l[v] * corr + rsum;
            m[v] = mnew;
            o[0][v] *= corr; o[1][v] *= corr; o[2][v] *= corr; o[3][v] *= corr;
            pl[v + 8 * laneHi][laneLo]      = f2bf(p0);
            pl[v + 8 * laneHi][16 + laneLo] = f2bf(p1);
        }

        // ---- P (C-layout) -> A-layout via LDS (same-wave DS is in-order) ----
        v16bf pa;
        {
            const __bf16* pr = &pl[laneLo][0];
            v8bf lo = *reinterpret_cast<const v8bf*>(pr + ka);
            v8bf hi = *reinterpret_cast<const v8bf*>(pr + ka + 8);
#pragma unroll
            for (int i = 0; i < 8; ++i) { pa[i] = lo[i]; pa[8 + i] = hi[i]; }
        }

        // ---- O += P[16x32] @ V[32x64], 4 WMMAs (Vt gives contiguous B) ----
#pragma unroll
        for (int t = 0; t < 4; ++t) {
            const __bf16* vp = vbaseP + (size_t)(t * 16 + laneLo) * SEQ
                                      + k0 + (laneHi ? 16 : 0);
            v16bf vb = *reinterpret_cast<const v16bf*>(vp);
            o[t] = wmma_bf16(pa, vb, o[t]);
        }
    }

    // ---- normalize and store to [b, s, h*64 + d] ----
    __bf16* aop = AO + ((size_t)(b * SEQ + qt * 16)) * HID + h * HD;
#pragma unroll
    for (int v = 0; v < 8; ++v) {
        float invl = 1.0f / l[v];
        int row = v + 8 * laneHi;
#pragma unroll
        for (int t = 0; t < 4; ++t)
            aop[(size_t)row * HID + t * 16 + laneLo] = f2bf(o[t][v] * invl);
    }
}

// ---------------------------------------------------------------------------
// Kernel 3: output projection.  A [4096,2048] bf16  x  WoT [2048,2048] bf16
//           -> out [4096,2048] f32
// ---------------------------------------------------------------------------
__global__ __launch_bounds__(32) void outproj_kernel(const __bf16* __restrict__ A,
                                                     const __bf16* __restrict__ WoT,
                                                     float* __restrict__ outp) {
    const int lane   = threadIdx.x & 31;
    const int laneLo = lane & 15;
    const int laneHi = lane >> 4;
    const int rtile  = blockIdx.x;           // 256
    const int nt     = blockIdx.y;           // 32 groups of 64 cols
    const int row    = rtile * 16 + laneLo;
    const int colBase = nt * 64;
    const int kaOff  = laneHi ? 16 : 0;
    const int kbOff  = laneHi ? 16 : 0;

    v8f acc[4];
#pragma unroll
    for (int t = 0; t < 4; ++t) acc[t] = vzero8();

    const __bf16* ar = A + (size_t)row * HID;
    for (int kb = 0; kb < HID; kb += 32) {
        v16bf a;
        v8bf lo = *reinterpret_cast<const v8bf*>(ar + kb + kaOff);
        v8bf hi = *reinterpret_cast<const v8bf*>(ar + kb + kaOff + 8);
#pragma unroll
        for (int i = 0; i < 8; ++i) { a[i] = lo[i]; a[8 + i] = hi[i]; }
#pragma unroll
        for (int t = 0; t < 4; ++t) {
            const __bf16* wp = WoT + (size_t)(colBase + t * 16 + laneLo) * HID
                                   + kb + kbOff;
            v16bf bm = *reinterpret_cast<const v16bf*>(wp);
            acc[t] = wmma_bf16(a, bm, acc[t]);
        }
    }

#pragma unroll
    for (int v = 0; v < 8; ++v) {
        int gr = rtile * 16 + v + 8 * laneHi;
#pragma unroll
        for (int t = 0; t < 4; ++t)
            outp[(size_t)gr * HID + colBase + t * 16 + laneLo] = acc[t][v];
    }
}

// ---------------------------------------------------------------------------
extern "C" void kernel_launch(void* const* d_in, const int* in_sizes, int n_in,
                              void* d_out, int out_size, void* d_ws, size_t ws_size,
                              hipStream_t stream) {
    (void)in_sizes; (void)n_in; (void)out_size; (void)ws_size;
    const float* X  = (const float*)d_in[0];
    // d_in[1] = causal mask: recomputed analytically inside attn_kernel
    const float* Wq = (const float*)d_in[2];
    const float* Wk = (const float*)d_in[3];
    const float* Wv = (const float*)d_in[4];
    const float* Wo = (const float*)d_in[5];
    float* out = (float*)d_out;

    char* ws = (char*)d_ws;
    __bf16* Qr  = (__bf16*)(ws);                         // 16 MB  [b,32,s,64]
    __bf16* Kr  = (__bf16*)(ws + 16777216);              //  4 MB  [b,8,s,64]
    __bf16* Vt  = (__bf16*)(ws + 20971520);              //  4 MB  [b,8,64,s]
    __bf16* AO  = (__bf16*)(ws + 25165824);              // 16 MB  [b,s,2048]
    __bf16* WqT = (__bf16*)(ws + 41943040);              //  8 MB  [2048,2048]
    __bf16* WkT = (__bf16*)(ws + 50331648);              //  2 MB  [512,2048]
    __bf16* WvT = (__bf16*)(ws + 52428800);              //  2 MB  [512,2048]
    __bf16* WoT = (__bf16*)(ws + 54525952);              //  8 MB  [2048,2048]

    dim3 wv(32, 1, 1);

    // 0) transpose+convert weights to bf16 (B-matrix friendly layout)
    wtc_kernel<<<(HID * HID + 255) / 256, 256, 0, stream>>>(Wq, WqT, HID, HID);
    wtc_kernel<<<(HID * 512 + 255) / 256, 256, 0, stream>>>(Wk, WkT, HID, 512);
    wtc_kernel<<<(HID * 512 + 255) / 256, 256, 0, stream>>>(Wv, WvT, HID, 512);
    wtc_kernel<<<(HID * HID + 255) / 256, 256, 0, stream>>>(Wo, WoT, HID, HID);

    // 1) QKV projections (+RoPE on Q,K; V stored transposed)
    proj_kernel<<<dim3(256, NH, 1),  wv, 0, stream>>>(X, WqT, Qr, 1, 0);
    proj_kernel<<<dim3(256, NKV, 1), wv, 0, stream>>>(X, WkT, Kr, 1, 0);
    proj_kernel<<<dim3(256, NKV, 1), wv, 0, stream>>>(X, WvT, Vt, 0, 1);

    // 2) causal flash attention (GQA)
    attn_kernel<<<dim3(128, NH, 2), wv, 0, stream>>>(Qr, Kr, Vt, AO);

    // 3) output projection -> f32
    outproj_kernel<<<dim3(256, 32, 1), wv, 0, stream>>>(AO, WoT, out);
}
